// NTXentLoss_50268297232759
// MI455X (gfx1250) — compile-verified, matrix operand order checked by reading, not run on hardware
//
#include <hip/hip_runtime.h>
#include <hip/hip_bf16.h>
#include <math.h>

// ---------------------------------------------------------------------------
// NT-Xent loss, fused.  B=4096, D=256, N=8192, T=0.5.
//   logits = (F F^T)/T, F = row-normalized [z_i; z_j], logits in [-2,2],
//   diag == 2 exactly  =>  fixed-max softmax:  S_r = sum_j exp(l_rj - 2),
//   masked sum = S_r - 1,  nll_r = log(S_r - 1) + 2 - l_{r,pos}, pos = r^4096,
//   loss = sum_r nll_r / N^2.
// ---------------------------------------------------------------------------

#define NTX_B 4096
#define NTX_N 8192
#define NTX_D 256

#define NCHUNK   8                      // column split (deterministic partials)
#define CHUNK    (NTX_N / NCHUNK)       // 1024 columns per block
#define STG_COLS 64                     // columns per LDS stage
#define NSTG     (CHUNK / STG_COLS)     // 16 stages
#define STG_HALF (STG_COLS * NTX_D)     // halfs per stage buffer
#define STG_B    (STG_HALF * 2)         // 32 KB per buffer

typedef __attribute__((ext_vector_type(16))) _Float16 v16h;
typedef __attribute__((ext_vector_type(8)))  _Float16 v8h;
typedef __attribute__((ext_vector_type(8)))  float    v8f;

union ABfrag { v16h v; v8h h[2]; };

// ---------------------------------------------------------------------------
// Kernel 1: L2-normalize each row, store as f16.  One wave32 per row.
// ---------------------------------------------------------------------------
__global__ void ntx_normalize(const float* __restrict__ zi,
                              const float* __restrict__ zj,
                              _Float16* __restrict__ feats) {
    const int lane = threadIdx.x & 31;
    const int row  = (blockIdx.x * blockDim.x + threadIdx.x) >> 5;
    if (row >= NTX_N) return;

    const float* src = (row < NTX_B) ? (zi + (size_t)row * NTX_D)
                                     : (zj + (size_t)(row - NTX_B) * NTX_D);
    float v[8];
    float ss = 0.0f;
#pragma unroll
    for (int j = 0; j < 8; ++j) {
        v[j] = src[j * 32 + lane];
        ss += v[j] * v[j];
    }
#pragma unroll
    for (int m = 1; m < 32; m <<= 1) ss += __shfl_xor(ss, m, 32);
    const float inv = rsqrtf(ss);

    _Float16* dst = feats + (size_t)row * NTX_D;
#pragma unroll
    for (int j = 0; j < 8; ++j) dst[j * 32 + lane] = (_Float16)(v[j] * inv);
}

// ---------------------------------------------------------------------------
// Kernel 2: partial rowsums  P[ck][r] = sum_{j in chunk ck} exp(2*dot - 2).
// Block = 8 waves x 16 rows (A register-resident), column chunk of 1024
// swept through a double-buffered 2x32KB LDS stage filled with
// global_load_async_to_lds_b128 (ASYNCcnt), consumed via ds_load -> WMMA.
// ---------------------------------------------------------------------------
__global__ void __launch_bounds__(256)
ntx_gemm_rowsum(const _Float16* __restrict__ feats,
                float* __restrict__ partial) {
    __shared__ _Float16 sbuf[2 * STG_HALF];     // 64 KB

    const int tid   = threadIdx.x;
    const int lane  = tid & 31;
    const int wave  = tid >> 5;
    const int mbase = blockIdx.x * 128 + wave * 16;   // 64 row-blocks
    const int cbase = blockIdx.y * CHUNK;             // 8 column chunks
    const int lrow  = lane & 15;
    const int kh    = lane >> 4;                      // K-half this lane holds

    // LDS byte offset of sbuf (addrspace(3) cast -> raw LDS offset)
    typedef __attribute__((address_space(3))) _Float16 lds_half_t;
    const unsigned lds_base =
        (unsigned)(unsigned long long)(lds_half_t*)&sbuf[0];

    // A fragments: 16 rows x 256 K in registers (8 x v16h), loaded once.
    ABfrag a[8];
    {
        const _Float16* ar = feats + (size_t)(mbase + lrow) * NTX_D;
#pragma unroll
        for (int kc = 0; kc < 8; ++kc) {
            a[kc].h[0] = *(const v8h*)(ar + kc * 32 + kh * 8);
            a[kc].h[1] = *(const v8h*)(ar + kc * 32 + 16 + kh * 8);
        }
    }

    float rs[8];
#pragma unroll
    for (int i = 0; i < 8; ++i) rs[i] = 0.0f;

    // Stage s: linear 32 KB copy of rows [cbase+s*64, +64) into sbuf half
    // `buf`; 2048 b128 transfers, 8 per thread, via async global->LDS.
    auto stage = [&](int buf, int s) {
        const _Float16* gbase = feats + (size_t)(cbase + s * STG_COLS) * NTX_D;
#pragma unroll
        for (int i = 0; i < 8; ++i) {
            const int q = tid + i * 256;                       // 0..2047
            const unsigned loff = lds_base + (unsigned)buf * STG_B + q * 16u;
            const _Float16* g = gbase + q * 8;                 // q*16 bytes
            asm volatile("global_load_async_to_lds_b128 %0, %1, off"
                         :: "v"(loff), "v"(g) : "memory");
        }
    };

    stage(0, 0);
    for (int s = 0; s < NSTG; ++s) {
        const int buf = s & 1;
        if (s + 1 < NSTG) {
            stage(buf ^ 1, s + 1);
            asm volatile("s_wait_asynccnt 0x8" ::: "memory"); // stage s done
        } else {
            asm volatile("s_wait_asynccnt 0x0" ::: "memory");
        }
        __syncthreads();

        const _Float16* bb = sbuf + buf * STG_HALF;
#pragma unroll
        for (int tt = 0; tt < 4; ++tt) {                      // 4 tiles of 16
            const _Float16* br = bb + (size_t)(tt * 16 + lrow) * NTX_D;
            ABfrag b[8];
#pragma unroll
            for (int kc = 0; kc < 8; ++kc) {
                b[kc].h[0] = *(const v8h*)(br + kc * 32 + kh * 8);
                b[kc].h[1] = *(const v8h*)(br + kc * 32 + 16 + kh * 8);
            }
            v8f acc = {};
#pragma unroll
            for (int kc = 0; kc < 8; ++kc) {
                acc = __builtin_amdgcn_wmma_f32_16x16x32_f16(
                    false, a[kc].v, false, b[kc].v,
                    (short)0, acc, false, false);
            }
            // exp(logit - 2) = exp(2*dot - 2), exponent in [-4,0]
#pragma unroll
            for (int i = 0; i < 8; ++i)
                rs[i] += __expf(2.0f * acc[i] - 2.0f);
        }
        __syncthreads();   // buffer `buf` free for reuse next-next stage
    }

    // C layout: VGPR i -> row mbase+i (+8 for lanes 16-31), col = lane&15.
#pragma unroll
    for (int m = 1; m < 16; m <<= 1) {
#pragma unroll
        for (int i = 0; i < 8; ++i) rs[i] += __shfl_xor(rs[i], m, 32);
    }
    if (lrow == 0) {
        float* out = partial + (size_t)blockIdx.y * NTX_N + mbase + kh * 8;
#pragma unroll
        for (int i = 0; i < 8; ++i) out[i] = rs[i];
    }
}

// ---------------------------------------------------------------------------
// Kernel 3: combine column partials, positive logit, per-row nll.
// ---------------------------------------------------------------------------
__global__ void ntx_nll(const _Float16* __restrict__ feats,
                        const float* __restrict__ partial,
                        float* __restrict__ nll) {
    const int lane = threadIdx.x & 31;
    const int row  = (blockIdx.x * blockDim.x + threadIdx.x) >> 5;
    if (row >= NTX_N) return;
    const int pos = row ^ NTX_B;

    const _Float16* fr = feats + (size_t)row * NTX_D;
    const _Float16* fp = feats + (size_t)pos * NTX_D;
    float d = 0.0f;
#pragma unroll
    for (int j = 0; j < 8; ++j)
        d += (float)fr[j * 32 + lane] * (float)fp[j * 32 + lane];
#pragma unroll
    for (int m = 1; m < 32; m <<= 1) d += __shfl_xor(d, m, 32);

    if (lane == 0) {
        float S = 0.0f;
#pragma unroll
        for (int c = 0; c < NCHUNK; ++c) S += partial[(size_t)c * NTX_N + row];
        // masked logsumexp = log(S - 1) + 2 ;  nll = that - 2*dot_pos
        nll[row] = logf(S - 1.0f) + 2.0f - 2.0f * d;
    }
}

// ---------------------------------------------------------------------------
// Kernel 4: deterministic single-block reduction; loss = sum(nll)/N^2.
// ---------------------------------------------------------------------------
__global__ void ntx_reduce(const float* __restrict__ nll,
                           float* __restrict__ out) {
    __shared__ float sm[256];
    const int t = threadIdx.x;
    float s = 0.0f;
    for (int i = t; i < NTX_N; i += 256) s += nll[i];
    sm[t] = s;
    __syncthreads();
    for (int off = 128; off > 0; off >>= 1) {
        if (t < off) sm[t] += sm[t + off];
        __syncthreads();
    }
    if (t == 0)
        out[0] = sm[0] * (1.0f / ((float)NTX_N * (float)NTX_N));
}

// ---------------------------------------------------------------------------
extern "C" void kernel_launch(void* const* d_in, const int* in_sizes, int n_in,
                              void* d_out, int out_size, void* d_ws, size_t ws_size,
                              hipStream_t stream) {
    (void)in_sizes; (void)n_in; (void)out_size; (void)ws_size;
    const float* zi = (const float*)d_in[0];
    const float* zj = (const float*)d_in[1];
    float* out = (float*)d_out;

    // Workspace layout (~4.3 MB; feats sit entirely in the 192 MB L2).
    char* ws = (char*)d_ws;
    _Float16* feats   = (_Float16*)(ws);                          // 4 MB
    float*    partial = (float*)(ws + (size_t)NTX_N * NTX_D * 2); // 256 KB
    float*    nll     = partial + (size_t)NCHUNK * NTX_N;         // 32 KB

    // 1) normalize: 8192 waves -> 1024 blocks x 8 waves
    ntx_normalize<<<NTX_N / 8, 256, 0, stream>>>(zi, zj, feats);
    // 2) fused GEMM + exp rowsum partials: 512 blocks = 64 row x 8 col chunks
    ntx_gemm_rowsum<<<dim3(64, NCHUNK), 256, 0, stream>>>(feats, partial);
    // 3) combine partials + positive logit -> per-row nll
    ntx_nll<<<NTX_N / 8, 256, 0, stream>>>(feats, partial, nll);
    // 4) final deterministic reduction
    ntx_reduce<<<1, 256, 0, stream>>>(nll, out);
}
